// NeuralODE_29403346108771
// MI455X (gfx1250) — matrix operand order repaired and checked by reading, not compiled, not run on hardware
//
#include <hip/hip_runtime.h>
#include <hip/hip_bf16.h>

typedef __attribute__((ext_vector_type(16))) _Float16 v16h;
typedef __attribute__((ext_vector_type(8)))  _Float16 v8h;
typedef __attribute__((ext_vector_type(8)))  float    v8f;

#define DEVINL __device__ __forceinline__

// ---------------- Tsit5 tableau ----------------
__constant__ float ACOEF[6][5] = {
    {0.f, 0.f, 0.f, 0.f, 0.f},
    {0.161f, 0.f, 0.f, 0.f, 0.f},
    {-0.008480655492356989f, 0.335480655492357f, 0.f, 0.f, 0.f},
    {2.8971530571054935f, -6.359448489975075f, 4.3622954328695815f, 0.f, 0.f},
    {5.325864828439257f, -11.748883564062828f, 7.4955393428898365f, -0.09249506636175525f, 0.f},
    {5.86145544294642f, -12.92096931784711f, 8.159367898576159f, -0.071584973281401f, -0.028269050394068383f}};
__constant__ float BCOEF[6] = {
    0.09646076681806523f, 0.01f, 0.4798896504144996f,
    1.379008574103742f, -3.290069515436081f, 2.324710524099774f};

// ---------------- scalar helpers ----------------
DEVINL float fast_tanh(float x) {
#if __has_builtin(__builtin_amdgcn_tanhf)
  return __builtin_amdgcn_tanhf(x);   // v_tanh_f32 (gfx1250 TRANS op)
#elif __has_builtin(__builtin_amdgcn_tanh_f32)
  return __builtin_amdgcn_tanh_f32(x);
#else
  return tanhf(x);
#endif
}

DEVINL float fast_sigmoid(float x) { return 1.f / (1.f + __expf(-x)); }

// ---------------- WMMA helpers ----------------
DEVINL v8f wmma_f16(v16h a, v16h b, v8f c) {
  // D = A(16x32 f16) * B(32x16 f16) + C(16x16 f32)
  return __builtin_amdgcn_wmma_f32_16x16x32_f16(false, a, false, b, (short)0, c,
                                                false, false);
}

DEVINL v8f zero8() {
  v8f z;
#pragma unroll
  for (int i = 0; i < 8; ++i) z[i] = 0.f;
  return z;
}

DEVINL v8f splat8(float v) {
  v8f z;
#pragma unroll
  for (int i = 0; i < 8; ++i) z[i] = v;
  return z;
}

// A-fragment (16x32 f16) from row-major [M][K] storage, ld in elements.
// Lane L holds M = L%16; K pattern: lanes<16 -> {0..7,16..23}, lanes>=16 -> {8..15,24..31}.
DEVINL v16h frag_a_f16(const _Float16* src, int ld, int row0, int k0, int lane) {
  const _Float16* p = src + (size_t)(row0 + (lane & 15)) * ld + k0 + ((lane >> 4) << 3);
  v8h lo = *(const v8h*)p;
  v8h hi = *(const v8h*)(p + 16);
  return __builtin_shufflevector(lo, hi, 0, 1, 2, 3, 4, 5, 6, 7, 8, 9, 10, 11, 12, 13, 14, 15);
}

// B-fragment (32x16 f16) of W^T where W is stored row-major [N][K].
// Lane L holds N = L%16; K pattern: lanes<16 -> {0..7,16..23}, lanes>=16 -> {8..15,24..31}.
DEVINL v16h frag_b_f16(const _Float16* w, int ld, int n0, int k0, int lane) {
  const _Float16* p = w + (size_t)(n0 + (lane & 15)) * ld + k0 + ((lane >> 4) << 3);
  v8h lo = *(const v8h*)p;
  v8h hi = *(const v8h*)(p + 16);
  return __builtin_shufflevector(lo, hi, 0, 1, 2, 3, 4, 5, 6, 7, 8, 9, 10, 11, 12, 13, 14, 15);
}

// A-fragment built from f32 source with on-the-fly convert (full K=32 tile).
DEVINL v16h frag_a_f32(const float* src, int ld, int row0, int k0, int lane) {
  const float* p = src + (size_t)(row0 + (lane & 15)) * ld + k0;
  int kh = (lane >> 4) << 3;
  v16h f;
#pragma unroll
  for (int e = 0; e < 8; ++e) f[e] = (_Float16)p[kh + e];
#pragma unroll
  for (int e = 0; e < 8; ++e) f[8 + e] = (_Float16)p[16 + kh + e];
  return f;
}

// A-fragment for a K=16 f32 tile zero-padded to K=32 (GRU input projection).
// Valid K: lanes<16 -> elements 0..7 = K0..7 ; lanes>=16 -> elements 0..7 = K8..15.
// Elements 8..15 (K=16..31) are zero.
DEVINL v16h frag_a_x16(const float* xs, int row0, int lane) {
  const float* p = xs + (size_t)(row0 + (lane & 15)) * 16 + ((lane >> 4) << 3);
  v16h f;
#pragma unroll
  for (int e = 0; e < 8; ++e) f[e] = (_Float16)p[e];
#pragma unroll
  for (int e = 0; e < 8; ++e) f[8 + e] = (_Float16)0.f;
  return f;
}

// One 256-wide hidden layer: acc[8] covers tiles (m0, (ntb+i)*16), K=256.
// Prefetch next K-slice of the (L2-resident) weight matrix while computing.
DEVINL void gemm256_acc(const _Float16* xin, const _Float16* w, v8f* acc,
                        int m0, int ntb, int lane) {
#pragma unroll
  for (int kk = 0; kk < 8; ++kk) {
    if (kk + 1 < 8)
      __builtin_prefetch(w + (size_t)(ntb * 16 + (lane & 15)) * 256 +
                             (kk + 1) * 32 + ((lane >> 4) << 4), 0, 0);
    v16h a = frag_a_f16(xin, 256, m0, kk * 32, lane);
#pragma unroll
    for (int i = 0; i < 8; ++i)
      acc[i] = wmma_f16(a, frag_b_f16(w, 256, (ntb + i) * 16, kk * 32, lane), acc[i]);
  }
}

DEVINL void write_tiles_f16(_Float16* xout, const v8f* acc, int m0, int ntb,
                            int lane, bool do_tanh) {
  int n = lane & 15, mh = (lane >> 4) << 3;
#pragma unroll
  for (int i = 0; i < 8; ++i) {
#pragma unroll
    for (int r = 0; r < 8; ++r) {
      float v = acc[i][r];
      xout[(size_t)(m0 + mh + r) * 256 + (ntb + i) * 16 + n] =
          (_Float16)(do_tanh ? fast_tanh(v) : v);
    }
  }
}

// ---------------- CDNA5 async global->LDS copy (ASYNCcnt path) ----------------
// Each thread copies 16 bytes: 1024 threads move a [256,16] f32 tile (16 KB).
DEVINL void async_load_x_tile(const float* __restrict__ yi, int t,
                              float* xstage, int tid) {
  int row = tid >> 2, c4 = (tid & 3) << 2;  // 4 dwords per thread
  const float* gsrc = yi + ((size_t)row * 256 + t) * 16 + c4;
  unsigned lds_addr = (unsigned)(uintptr_t)(xstage + row * 16 + c4);
  unsigned long long gaddr = (unsigned long long)(uintptr_t)gsrc;
  asm volatile("global_load_async_to_lds_b128 %0, %1, off"
               :
               : "v"(lds_addr), "v"(gaddr)
               : "memory");
}

// ---------------- f32 -> f16 weight packing ----------------
__global__ void pack_f16_kernel(const float* __restrict__ src,
                                _Float16* __restrict__ dst, int n) {
  int i = blockIdx.x * blockDim.x + threadIdx.x;
  if (i < n) dst[i] = (_Float16)src[i];
}

// ---------------- GRU encoder scan (single persistent workgroup) ----------------
// h_{t} over reversed sequence; B=256, H=128, 3H=384, D=16 (padded to K=32).
// x-tiles are double-buffered via async global->LDS DMA.
__global__ __launch_bounds__(1024) void gru_scan_kernel(
    const float* __restrict__ yi, const float* __restrict__ wih,
    const float* __restrict__ whh, const float* __restrict__ bias,
    const float* __restrict__ bn, float* __restrict__ hidden_out) {
  extern __shared__ char smem[];
  _Float16* whh_sm = (_Float16*)smem;          // 384*128
  _Float16* wih_sm = whh_sm + 384 * 128;       // 384*32 (K padded 16->32)
  _Float16* hA = wih_sm + 384 * 32;            // 256*128
  _Float16* hB = hA + 256 * 128;               // 256*128
  float* xs0 = (float*)(hB + 256 * 128);       // 256*16 f32 (async buf 0)
  float* xs1 = xs0 + 256 * 16;                 // 256*16 f32 (async buf 1)
  float* b_sm = xs1 + 256 * 16;                // 384
  float* bn_sm = b_sm + 384;                   // 128

  const int tid = threadIdx.x, wv = tid >> 5, lane = tid & 31;

  for (int i = tid; i < 384 * 128; i += 1024) whh_sm[i] = (_Float16)whh[i];
  for (int i = tid; i < 384 * 32; i += 1024) {
    int r = i >> 5, c = i & 31;
    wih_sm[i] = (_Float16)((c < 16) ? wih[r * 16 + c] : 0.f);
  }
  for (int i = tid; i < 256 * 128; i += 1024) hA[i] = (_Float16)0.f;
  for (int i = tid; i < 384; i += 1024) b_sm[i] = bias[i];
  for (int i = tid; i < 128; i += 1024) bn_sm[i] = bn[i];

  // prologue: kick off async DMA of the first (t=255) x tile
  async_load_x_tile(yi, 255, xs0, tid);
  __syncthreads();

  _Float16* hcur = hA;
  _Float16* hnext = hB;
  for (int s = 0; s < 256; ++s) {
    int t = 255 - s;  // reversed sequence
    const float* xcur = (s & 1) ? xs1 : xs0;
    if (s < 255) {
      float* xnext = (s & 1) ? xs0 : xs1;
      async_load_x_tile(yi, t - 1, xnext, tid);  // overlap with this step
      asm volatile("s_wait_asynccnt 0x1" ::: "memory");  // wait current tile
    } else {
      asm volatile("s_wait_asynccnt 0x0" ::: "memory");
    }
    __syncthreads();  // all waves' tile portions visible

#pragma unroll
    for (int q = 0; q < 4; ++q) {
      int tile = wv * 4 + q;
      int mt = tile >> 3, jt = tile & 7, m0 = mt * 16;
      int n = lane & 15, mh = (lane >> 4) << 3, col = jt * 16 + n;

      v8f accR = splat8(b_sm[col]);
      v8f accZ = splat8(b_sm[128 + col]);
      v8f accC = splat8(b_sm[256 + col]);
      v8f accN = zero8();

      // input gates: K=32 (K16 zero-padded) single WMMA each
      v16h ax = frag_a_x16(xcur, m0, lane);
      accR = wmma_f16(ax, frag_b_f16(wih_sm, 32, jt * 16, 0, lane), accR);
      accZ = wmma_f16(ax, frag_b_f16(wih_sm, 32, 128 + jt * 16, 0, lane), accZ);
      accC = wmma_f16(ax, frag_b_f16(wih_sm, 32, 256 + jt * 16, 0, lane), accC);
      // hidden gates: K=128
#pragma unroll
      for (int kk = 0; kk < 4; ++kk) {
        v16h ah = frag_a_f16(hcur, 128, m0, kk * 32, lane);
        accR = wmma_f16(ah, frag_b_f16(whh_sm, 128, jt * 16, kk * 32, lane), accR);
        accZ = wmma_f16(ah, frag_b_f16(whh_sm, 128, 128 + jt * 16, kk * 32, lane), accZ);
        accN = wmma_f16(ah, frag_b_f16(whh_sm, 128, 256 + jt * 16, kk * 32, lane), accN);
      }

      float bnv = bn_sm[col];
#pragma unroll
      for (int r = 0; r < 8; ++r) {
        int row = m0 + mh + r;
        float hp = (float)hcur[row * 128 + col];
        float rg = fast_sigmoid(accR[r]);
        float zg = fast_sigmoid(accZ[r]);
        float ng = fast_tanh(accC[r] + rg * (accN[r] + bnv));
        hnext[row * 128 + col] = (_Float16)(ng + zg * (hp - ng));
      }
    }
    __syncthreads();
    _Float16* tmp = hcur; hcur = hnext; hnext = tmp;
  }

  for (int i = tid; i < 256 * 128; i += 1024) hidden_out[i] = (float)hcur[i];
}

// ---------------- hidden -> y0 MLP (tanh, tanh, linear) ----------------
__global__ __launch_bounds__(1024) void h2o_kernel(
    const float* __restrict__ hidden, const _Float16* __restrict__ w1,
    const float* __restrict__ b1, const _Float16* __restrict__ w2,
    const float* __restrict__ b2, const _Float16* __restrict__ w3,
    const float* __restrict__ b3, float* __restrict__ ystate,
    float* __restrict__ ys) {
  extern __shared__ char smem[];
  _Float16* hbuf = (_Float16*)smem;      // 256*128
  _Float16* xbuf = hbuf + 256 * 128;     // 256*256

  const int tid = threadIdx.x, wv = tid >> 5, lane = tid & 31;
  const int n = lane & 15, mh = (lane >> 4) << 3;
  const int m0 = (wv >> 1) * 16, ntb = (wv & 1) * 8;

  for (int i = tid; i < 256 * 128; i += 1024) hbuf[i] = (_Float16)hidden[i];
  __syncthreads();

  v8f acc[8];
  // layer1: [256,128] @ W1^T -> [256,256], tanh
#pragma unroll
  for (int i = 0; i < 8; ++i) acc[i] = splat8(b1[(ntb + i) * 16 + n]);
#pragma unroll
  for (int kk = 0; kk < 4; ++kk) {
    v16h a = frag_a_f16(hbuf, 128, m0, kk * 32, lane);
#pragma unroll
    for (int i = 0; i < 8; ++i)
      acc[i] = wmma_f16(a, frag_b_f16(w1, 128, (ntb + i) * 16, kk * 32, lane), acc[i]);
  }
  write_tiles_f16(xbuf, acc, m0, ntb, lane, true);
  __syncthreads();

  // layer2: [256,256] @ W2^T, tanh (regs -> barrier -> write back in place)
#pragma unroll
  for (int i = 0; i < 8; ++i) acc[i] = splat8(b2[(ntb + i) * 16 + n]);
  gemm256_acc(xbuf, w2, acc, m0, ntb, lane);
  __syncthreads();
  write_tiles_f16(xbuf, acc, m0, ntb, lane, true);
  __syncthreads();

  // layer3: [256,256] @ W3^T -> [256,32]; one tile per wave
  int no = wv & 1;
  v8f a3 = splat8(b3[no * 16 + n]);
#pragma unroll
  for (int kk = 0; kk < 8; ++kk)
    a3 = wmma_f16(frag_a_f16(xbuf, 256, m0, kk * 32, lane),
                  frag_b_f16(w3, 256, no * 16, kk * 32, lane), a3);
#pragma unroll
  for (int r = 0; r < 8; ++r) {
    int row = m0 + mh + r, col = no * 16 + n;
    float v = a3[r];
    ystate[row * 32 + col] = v;
    ys[row * 32 + col] = v;  // ys[t=0]
  }
}

// Build A-fragment of yin = y + h * sum_j A[s][j] * k_j  (f32 -> f16, K=32)
DEVINL v16h build_yin(const float* __restrict__ y, const float* __restrict__ kb,
                      int s, float h, int m0, int lane) {
  int m = m0 + (lane & 15);
  int kh = (lane >> 4) << 3;
  v16h f;
#pragma unroll
  for (int e = 0; e < 16; ++e) {
    int k = (e < 8) ? (kh + e) : (16 + kh + (e - 8));
    float v = y[m * 32 + k];
    for (int j = 0; j < s; ++j)
      v += h * ACOEF[s][j] * kb[(size_t)(j * 256 + m) * 32 + k];
    f[e] = (_Float16)v;
  }
  return f;
}

// ---------------- Tsit5 ODE scan (single persistent workgroup) ----------------
__global__ __launch_bounds__(1024) void ode_scan_kernel(
    const float* __restrict__ ts, const _Float16* __restrict__ w0,
    const _Float16* __restrict__ w1, const _Float16* __restrict__ w2,
    const _Float16* __restrict__ wout, float* __restrict__ ystate,
    float* __restrict__ ys) {
  extern __shared__ char smem[];
  _Float16* xbuf = (_Float16*)smem;           // 256*256 f16 = 128 KB
  float* kbuf = (float*)(xbuf + 256 * 256);   // 6*256*32 f32 = 192 KB

  const int tid = threadIdx.x, wv = tid >> 5, lane = tid & 31;
  const int n = lane & 15, mh = (lane >> 4) << 3;
  const int m0 = (wv >> 1) * 16, ntb = (wv & 1) * 8, no = wv & 1;

  for (int t = 1; t < 256; ++t) {
    float hstep = ts[t] - ts[t - 1];

    for (int s = 0; s < 6; ++s) {
      v8f acc[8];
      // layer0: yin[256,32] @ W0^T -> [256,256], tanh (bias-free)
      {
        v16h a = build_yin(ystate, kbuf, s, hstep, m0, lane);
#pragma unroll
        for (int i = 0; i < 8; ++i)
          acc[i] = wmma_f16(a, frag_b_f16(w0, 32, (ntb + i) * 16, 0, lane), zero8());
      }
      write_tiles_f16(xbuf, acc, m0, ntb, lane, true);
      __syncthreads();

      // layer1: [256,256] @ W1^T, tanh
#pragma unroll
      for (int i = 0; i < 8; ++i) acc[i] = zero8();
      gemm256_acc(xbuf, w1, acc, m0, ntb, lane);
      __syncthreads();
      write_tiles_f16(xbuf, acc, m0, ntb, lane, true);
      __syncthreads();

      // layer2: [256,256] @ W2^T, tanh
#pragma unroll
      for (int i = 0; i < 8; ++i) acc[i] = zero8();
      gemm256_acc(xbuf, w2, acc, m0, ntb, lane);
      __syncthreads();
      write_tiles_f16(xbuf, acc, m0, ntb, lane, true);
      __syncthreads();

      // layerOut: [256,256] @ Wout^T -> k_s [256,32]; one tile per wave
      v8f ao = zero8();
#pragma unroll
      for (int kk = 0; kk < 8; ++kk)
        ao = wmma_f16(frag_a_f16(xbuf, 256, m0, kk * 32, lane),
                      frag_b_f16(wout, 256, no * 16, kk * 32, lane), ao);
#pragma unroll
      for (int r = 0; r < 8; ++r) {
        int row = m0 + mh + r, col = no * 16 + n;
        kbuf[(size_t)(s * 256 + row) * 32 + col] = ao[r];
      }
      __syncthreads();
    }

    // y_{t} = y + h * sum_i B_i k_i ; save to trajectory
    for (int i = tid; i < 8192; i += 1024) {
      int m = i >> 5, k = i & 31;
      float v = ystate[i];
#pragma unroll
      for (int j = 0; j < 6; ++j)
        v += hstep * BCOEF[j] * kbuf[(size_t)(j * 256 + m) * 32 + k];
      ystate[i] = v;
      ys[(size_t)t * 8192 + i] = v;
    }
    __threadfence_block();
    __syncthreads();
  }
}

// ---------------- ODE -> data decoder (identity activations) ----------------
__global__ __launch_bounds__(1024) void o2d_kernel(
    const float* __restrict__ ys, const _Float16* __restrict__ w1,
    const float* __restrict__ b1, const _Float16* __restrict__ w2,
    const float* __restrict__ b2, const _Float16* __restrict__ w3,
    const float* __restrict__ b3, float* __restrict__ out) {
  extern __shared__ char smem[];
  _Float16* xbuf = (_Float16*)smem;  // 256*256

  const int tid = threadIdx.x, wv = tid >> 5, lane = tid & 31;
  const int n = lane & 15, mh = (lane >> 4) << 3;
  const int m0 = (wv >> 1) * 16, ntb = (wv & 1) * 8;
  const size_t r0 = (size_t)blockIdx.x * 256;  // flat row = t*256 + b
  const float* yrows = ys + r0 * 32;

  v8f acc[8];
  // layer1: [256,32] @ W1^T -> [256,256] + b1 (identity)
  {
    v16h a = frag_a_f32(yrows, 32, m0, 0, lane);
#pragma unroll
    for (int i = 0; i < 8; ++i) {
      v8f c = splat8(b1[(ntb + i) * 16 + n]);
      acc[i] = wmma_f16(a, frag_b_f16(w1, 32, (ntb + i) * 16, 0, lane), c);
    }
  }
  write_tiles_f16(xbuf, acc, m0, ntb, lane, false);
  __syncthreads();

  // layer2: [256,256] @ W2^T + b2 (identity)
#pragma unroll
  for (int i = 0; i < 8; ++i) acc[i] = splat8(b2[(ntb + i) * 16 + n]);
  gemm256_acc(xbuf, w2, acc, m0, ntb, lane);
  __syncthreads();
  write_tiles_f16(xbuf, acc, m0, ntb, lane, false);
  __syncthreads();

  // layer3: [256,256] @ W3^T -> [256,16] + b3 ; 16 tiles on waves 0..15
  if (wv < 16) {
    v8f a3 = splat8(b3[n]);
#pragma unroll
    for (int kk = 0; kk < 8; ++kk)
      a3 = wmma_f16(frag_a_f16(xbuf, 256, wv * 16, kk * 32, lane),
                    frag_b_f16(w3, 256, 0, kk * 32, lane), a3);
#pragma unroll
    for (int r = 0; r < 8; ++r) {
      size_t gr = r0 + (size_t)(wv * 16 + mh + r);  // flat row in [T,B]
      size_t tt = gr >> 8, bb = gr & 255;
      out[((bb << 8) + tt) * 16 + n] = a3[r];       // out[b][t][n]
    }
  }
}

// ---------------- host launcher ----------------
extern "C" void kernel_launch(void* const* d_in, const int* in_sizes, int n_in,
                              void* d_out, int out_size, void* d_ws,
                              size_t ws_size, hipStream_t stream) {
  (void)in_sizes; (void)n_in; (void)out_size; (void)ws_size;
  const float* ts      = (const float*)d_in[0];
  const float* yi      = (const float*)d_in[1];
  const float* gru_wih = (const float*)d_in[2];
  const float* gru_whh = (const float*)d_in[3];
  const float* gru_b   = (const float*)d_in[4];
  const float* gru_bn  = (const float*)d_in[5];
  const float* h2o_W1  = (const float*)d_in[6];
  const float* h2o_b1  = (const float*)d_in[7];
  const float* h2o_W2  = (const float*)d_in[8];
  const float* h2o_b2  = (const float*)d_in[9];
  const float* h2o_W3  = (const float*)d_in[10];
  const float* h2o_b3  = (const float*)d_in[11];
  const float* cf_W0   = (const float*)d_in[12];
  const float* cf_W1   = (const float*)d_in[13];
  const float* cf_W2   = (const float*)d_in[14];
  const float* cf_Wout = (const float*)d_in[15];
  const float* o2d_W1  = (const float*)d_in[16];
  const float* o2d_b1  = (const float*)d_in[17];
  const float* o2d_W2  = (const float*)d_in[18];
  const float* o2d_b2  = (const float*)d_in[19];
  const float* o2d_W3  = (const float*)d_in[20];
  const float* o2d_b3  = (const float*)d_in[21];

  char* ws = (char*)d_ws;
  size_t off = 0;
  auto carve = [&](size_t bytes) -> char* {
    char* p = ws + off;
    off += (bytes + 255) & ~(size_t)255;
    return p;
  };

  float* hidden = (float*)carve((size_t)256 * 128 * 4);
  float* ystate = (float*)carve((size_t)256 * 32 * 4);
  float* ys     = (float*)carve((size_t)256 * 256 * 32 * 4);  // [T][B][ODE]
  _Float16* p_h2oW1 = (_Float16*)carve((size_t)256 * 128 * 2);
  _Float16* p_h2oW2 = (_Float16*)carve((size_t)256 * 256 * 2);
  _Float16* p_h2oW3 = (_Float16*)carve((size_t)32 * 256 * 2);
  _Float16* p_cfW0  = (_Float16*)carve((size_t)256 * 32 * 2);
  _Float16* p_cfW1  = (_Float16*)carve((size_t)256 * 256 * 2);
  _Float16* p_cfW2  = (_Float16*)carve((size_t)256 * 256 * 2);
  _Float16* p_cfWo  = (_Float16*)carve((size_t)32 * 256 * 2);
  _Float16* p_o2dW1 = (_Float16*)carve((size_t)256 * 32 * 2);
  _Float16* p_o2dW2 = (_Float16*)carve((size_t)256 * 256 * 2);
  _Float16* p_o2dW3 = (_Float16*)carve((size_t)16 * 256 * 2);

  auto pack = [&](const float* s, _Float16* dst, int nel) {
    pack_f16_kernel<<<(nel + 255) / 256, 256, 0, stream>>>(s, dst, nel);
  };
  pack(h2o_W1, p_h2oW1, 256 * 128);
  pack(h2o_W2, p_h2oW2, 256 * 256);
  pack(h2o_W3, p_h2oW3, 32 * 256);
  pack(cf_W0, p_cfW0, 256 * 32);
  pack(cf_W1, p_cfW1, 256 * 256);
  pack(cf_W2, p_cfW2, 256 * 256);
  pack(cf_Wout, p_cfWo, 32 * 256);
  pack(o2d_W1, p_o2dW1, 256 * 32);
  pack(o2d_W2, p_o2dW2, 256 * 256);
  pack(o2d_W3, p_o2dW3, 16 * 256);

  size_t smem_gru =
      (size_t)(384 * 128 + 384 * 32 + 2 * 256 * 128) * 2 +  // f16 arrays
      (size_t)(2 * 256 * 16 + 384 + 128) * 4;               // f32 arrays
  gru_scan_kernel<<<1, 1024, smem_gru, stream>>>(yi, gru_wih, gru_whh, gru_b,
                                                 gru_bn, hidden);

  size_t smem_h2o = (size_t)(256 * 128 + 256 * 256) * 2;
  h2o_kernel<<<1, 1024, smem_h2o, stream>>>(hidden, p_h2oW1, h2o_b1, p_h2oW2,
                                            h2o_b2, p_h2oW3, h2o_b3, ystate, ys);

  size_t smem_ode = (size_t)256 * 256 * 2 + (size_t)6 * 256 * 32 * 4;  // 320 KB
  ode_scan_kernel<<<1, 1024, smem_ode, stream>>>(ts, p_cfW0, p_cfW1, p_cfW2,
                                                 p_cfWo, ystate, ys);

  size_t smem_o2d = (size_t)256 * 256 * 2;
  o2d_kernel<<<256, 1024, smem_o2d, stream>>>(ys, p_o2dW1, o2d_b1, p_o2dW2,
                                              o2d_b2, p_o2dW3, o2d_b3,
                                              (float*)d_out);
}
